// AttnBlock_3624952398763
// MI455X (gfx1250) — matrix operand order repaired and checked by reading, not compiled, not run on hardware
//
#include <hip/hip_runtime.h>
#include <hip/hip_bf16.h>

// ---------------------------------------------------------------------------
// AttnBlock (GroupNorm + 1x1 conv QKV + softmax attention + proj + residual)
// for MI455X / gfx1250.  All GEMMs use v_wmma_f32_16x16x32_bf16 (wave32).
//
// Shapes: B=2, C=512, N=16*16*16=4096.  ~86 GFLOP total; all intermediates
// (~42MB bf16) are L2-resident (192MB), so the design targets the WMMA pipe.
// Attention is two-pass flash (stats, then output) to avoid the 128MB score
// matrix.
//
// Data-path choices for CDNA5:
//  * Loop-invariant A strips (16 rows x 512 K, 16KB/wave) are staged in LDS
//    (320KB/WGP) -- A fragments come from ds_load_b128 with short latency,
//    leaving the vector-memory pipe exclusively for the streaming K/V
//    B-fragments (2-stage software pipelined + global_prefetch).
//  * Dynamic LDS (128KB/block = 8 waves x 16KB); per-wave regions need no
//    block barrier (DS ops are wave-order), fenced with s_wait_dscnt 0.
//
// WMMA NT convention used everywhere:
//   D[16M x 16N] += A[16M x 32K] * B[32K x 16N]
//   A source memory: row-major [M][K]  (lane m holds row m; K 0-15 for
//                    lanes 0-15, K 16-31 for lanes 16-31)
//   B source memory: row-major [N][K]  (lane n holds col n; K 0-15 for lanes
//                    0-15, K 16-31 for lanes 16-31)
//   D layout: VGPR r, lanes 0-15 -> (row r,   col lane)
//             VGPR r, lanes16-31 -> (row r+8, col lane-16)
// ---------------------------------------------------------------------------

typedef __bf16 bf16;
typedef __attribute__((ext_vector_type(16))) __bf16 v16bf;
typedef __attribute__((ext_vector_type(8)))  __bf16 v8bf;
typedef __attribute__((ext_vector_type(8)))  float  v8f;
typedef __attribute__((ext_vector_type(4)))  float  v4f;
typedef __attribute__((ext_vector_type(4)))  unsigned int v4u;

#define DEVINL static __device__ __forceinline__

#define Bn     2
#define Cc     512
#define Nn     4096
#define NGRP   32
#define CPG    16          // channels per group
#define SCALE  0.044194173824159216f   // 1/sqrt(512), folded into qT

#define STRIP_BYTES (16 * Cc * 2)      // one wave's A strip in LDS (16KB)

DEVINL v8f zero8() {
  v8f z = {0.f, 0.f, 0.f, 0.f, 0.f, 0.f, 0.f, 0.f};
  return z;
}

DEVINL v8f wmma_bf16(v16bf a, v16bf b, v8f c) {
  return __builtin_amdgcn_wmma_f32_16x16x32_bf16(
      /*neg_a=*/false, a, /*neg_b=*/false, b,
      /*c_mod=*/(short)0, c, /*reuse_a=*/false, /*reuse_b=*/false);
}

// A fragment: src points to &mem[m0*ld + k0]; memory row-major [M][K], K contig.
DEVINL v16bf load_a_frag(const bf16* src, int ld) {
  const int lane = threadIdx.x & 31;
  const int off  = (lane >> 4) << 4;            // hi half -> K+16
  const bf16* p  = src + (size_t)(lane & 15) * ld + off;
  return *(const v16bf*)p;                      // K k0+off .. +15
}

// B fragment: src points to &mem[n0*ld + k0]; memory row-major [N][K], K contig.
DEVINL v16bf load_b_frag(const bf16* src, int ld) {
  const int lane = threadIdx.x & 31;
  const bf16* p  = src + (size_t)(lane & 15) * ld + ((lane >> 4) << 4);
  return *(const v16bf*)p;                      // 32B: K k0..k0+15 (or +16..31)
}

DEVINL v4u pack8_bf16(v8f a) {
  v8bf h;
#pragma unroll
  for (int i = 0; i < 8; ++i) h[i] = (__bf16)a[i];
  return __builtin_bit_cast(v4u, h);
}

// Copy one wave's 16x512 bf16 A strip (16KB) global -> LDS; wave-private, so
// no block barrier needed (DS ops complete in wave order); fence DScnt so the
// following ds_loads observe the staged data.
DEVINL void stage_strip16(bf16* dst_lds, const bf16* src) {
  const int lane = threadIdx.x & 31;
  const v4u* s = (const v4u*)src;
  v4u* d = (v4u*)dst_lds;
#pragma unroll
  for (int t = 0; t < 32; ++t)                  // 1024 x 16B chunks / 32 lanes
    d[lane + 32 * t] = s[lane + 32 * t];
  asm volatile("s_wait_dscnt 0" ::: "memory");
}

// ---------------------------------------------------------------------------
// 1) fp32 weights -> bf16 (4 matrices at once)
// ---------------------------------------------------------------------------
__global__ void k_w2bf(const float* wq, const float* wk, const float* wv,
                       const float* wp, bf16* oq, bf16* ok, bf16* ov, bf16* op_) {
  int i = blockIdx.x * blockDim.x + threadIdx.x;
  if (i < Cc * Cc) {
    oq[i]  = (bf16)wq[i];
    ok[i]  = (bf16)wk[i];
    ov[i]  = (bf16)wv[i];
    op_[i] = (bf16)wp[i];
  }
}

// ---------------------------------------------------------------------------
// 2) GroupNorm statistics: one block per (batch, group); 16ch x 4096 = 64K elems
// ---------------------------------------------------------------------------
__global__ void __launch_bounds__(256) k_gn_stats(const float* x, float* mu,
                                                  float* rstd) {
  const int bg = blockIdx.x;                       // b*32+g, groups contiguous
  const float* p = x + (size_t)bg * (CPG * Nn);
  float s1 = 0.f, s2 = 0.f;
  for (int i = threadIdx.x; i < CPG * Nn; i += blockDim.x) {
    float v = p[i];
    s1 += v;
    s2 += v * v;
  }
  __shared__ float sh1[256], sh2[256];
  sh1[threadIdx.x] = s1;
  sh2[threadIdx.x] = s2;
  __syncthreads();
  for (int s = 128; s > 0; s >>= 1) {
    if (threadIdx.x < (unsigned)s) {
      sh1[threadIdx.x] += sh1[threadIdx.x + s];
      sh2[threadIdx.x] += sh2[threadIdx.x + s];
    }
    __syncthreads();
  }
  if (threadIdx.x == 0) {
    const float inv = 1.f / (float)(CPG * Nn);
    float m  = sh1[0] * inv;
    float va = sh2[0] * inv - m * m;
    mu[bg]   = m;
    rstd[bg] = __frsqrt_rn(va + 1e-6f);
  }
}

// ---------------------------------------------------------------------------
// 3) GroupNorm apply + affine + transpose -> hnT[b][n][c] (bf16)
// ---------------------------------------------------------------------------
__global__ void __launch_bounds__(256) k_gn_apply(const float* x, const float* gw,
                                                  const float* gb, const float* mu,
                                                  const float* rstd, bf16* hnT) {
  size_t i = (size_t)blockIdx.x * blockDim.x + threadIdx.x;   // < B*C*N
  int n = (int)(i % Nn);
  int c = (int)((i / Nn) % Cc);
  int b = (int)(i / ((size_t)Nn * Cc));
  int bg = b * NGRP + c / CPG;
  float y = (x[i] - mu[bg]) * rstd[bg] * gw[c] + gb[c];
  hnT[((size_t)b * Nn + n) * Cc + c] = (bf16)y;
}

// ---------------------------------------------------------------------------
// 4) Q & K projection:  D[M=o][N=n] = W[o][c] * hnT[n][c]^T  (+bias)
//    16-token tile staged in LDS (B fragments shared by 8 waves); per c-step
//    all 8 A fragments are loaded as one batch (clause) before the 8 WMMAs,
//    and the next B fragment is prefetched from LDS.
//    qT additionally absorbs the 1/sqrt(C) attention scale.
// ---------------------------------------------------------------------------
__global__ void __launch_bounds__(256)
__attribute__((amdgpu_waves_per_eu(1, 4)))
k_proj_qk(const bf16* hnT, const bf16* wqb, const bf16* wkb, const float* bq,
          const float* bk, bf16* qT, bf16* kT) {
  const int b  = blockIdx.y;
  const int n0 = blockIdx.x * 16;
  __shared__ __align__(16) bf16 sh[16 * Cc];            // 16KB token tile
  const bf16* src = hnT + ((size_t)b * Nn + n0) * Cc;   // 16 contiguous rows
  for (int i = threadIdx.x; i < 16 * Cc / 8; i += blockDim.x)
    ((v4u*)sh)[i] = ((const v4u*)src)[i];
  __syncthreads();

  const int wave = threadIdx.x >> 5, lane = threadIdx.x & 31;
  const int hi = lane >> 4, cl = lane & 15;

  v8f accq[4], acck[4];
#pragma unroll
  for (int s = 0; s < 4; ++s) { accq[s] = zero8(); acck[s] = zero8(); }

  v16bf bfrag = load_b_frag(sh, Cc);
#pragma unroll
  for (int c0 = 0; c0 < Cc; c0 += 32) {
    v16bf af[8];                                        // batch all A loads
#pragma unroll
    for (int s = 0; s < 4; ++s) {
      const int o0 = wave * 64 + s * 16;
      af[2 * s]     = load_a_frag(wqb + (size_t)o0 * Cc + c0, Cc);
      af[2 * s + 1] = load_a_frag(wkb + (size_t)o0 * Cc + c0, Cc);
    }
    v16bf bnext;
    if (c0 + 32 < Cc) bnext = load_b_frag(sh + c0 + 32, Cc);
#pragma unroll
    for (int s = 0; s < 4; ++s) {
      accq[s] = wmma_bf16(af[2 * s],     bfrag, accq[s]);
      acck[s] = wmma_bf16(af[2 * s + 1], bfrag, acck[s]);
    }
    if (c0 + 32 < Cc) bfrag = bnext;
  }
#pragma unroll
  for (int s = 0; s < 4; ++s) {
    const int o0 = wave * 64 + s * 16 + 8 * hi;         // 8 consecutive o rows
    v8f q = accq[s], k = acck[s];
#pragma unroll
    for (int r = 0; r < 8; ++r) {
      q[r] = (q[r] + bq[o0 + r]) * SCALE;               // fold attention scale
      k[r] += bk[o0 + r];
    }
    bf16* dq = qT + ((size_t)b * Nn + n0 + cl) * Cc + o0;
    bf16* dk = kT + ((size_t)b * Nn + n0 + cl) * Cc + o0;
    *(v4u*)dq = pack8_bf16(q);
    *(v4u*)dk = pack8_bf16(k);
  }
}

// ---------------------------------------------------------------------------
// 5) V projection:  D[M=n][N=o] = hnT[n][c] * Wv[o][c]^T  (+bias)
//    Per-wave A strip staged in LDS; two o-tiles per A fragment.
//    Stored as v[c][n] so the attention O-GEMM B fragments are j-contiguous.
// ---------------------------------------------------------------------------
__global__ void __launch_bounds__(256)
__attribute__((amdgpu_waves_per_eu(1, 4)))
k_proj_v(const bf16* hnT, const bf16* wvb, const float* bv, bf16* vbuf) {
  extern __shared__ __align__(16) char smem_dyn[];
  const int b = blockIdx.y;
  const int wave = threadIdx.x >> 5, lane = threadIdx.x & 31;
  const int hi = lane >> 4, cl = lane & 15;
  const int n0 = blockIdx.x * 128 + wave * 16;

  bf16* ash = (bf16*)(smem_dyn + (size_t)wave * STRIP_BYTES);
  stage_strip16(ash, hnT + ((size_t)b * Nn + n0) * Cc);

  for (int o0 = 0; o0 < Cc; o0 += 32) {
    v8f acc0 = zero8(), acc1 = zero8();
    v16bf a  = load_a_frag(ash, Cc);
    v16bf b0 = load_b_frag(wvb + (size_t)o0 * Cc, Cc);
    v16bf b1 = load_b_frag(wvb + (size_t)(o0 + 16) * Cc, Cc);
#pragma unroll
    for (int c0 = 0; c0 < Cc; c0 += 32) {
      v16bf an, b0n, b1n;
      if (c0 + 32 < Cc) {
        an  = load_a_frag(ash + c0 + 32, Cc);
        b0n = load_b_frag(wvb + (size_t)o0 * Cc + c0 + 32, Cc);
        b1n = load_b_frag(wvb + (size_t)(o0 + 16) * Cc + c0 + 32, Cc);
      }
      acc0 = wmma_bf16(a, b0, acc0);
      acc1 = wmma_bf16(a, b1, acc1);
      if (c0 + 32 < Cc) { a = an; b0 = b0n; b1 = b1n; }
    }
    const float bias0 = bv[o0 + cl];
    const float bias1 = bv[o0 + 16 + cl];
#pragma unroll
    for (int r = 0; r < 8; ++r) { acc0[r] += bias0; acc1[r] += bias1; }
    // lane owns column o, rows n0+8*hi .. +7 -> contiguous in v[c][n]
    bf16* d0 = vbuf + ((size_t)b * Cc + o0 + cl) * Nn + n0 + 8 * hi;
    bf16* d1 = vbuf + ((size_t)b * Cc + o0 + 16 + cl) * Nn + n0 + 8 * hi;
    *(v4u*)d0 = pack8_bf16(acc0);
    *(v4u*)d1 = pack8_bf16(acc1);
  }
}

// ---------------------------------------------------------------------------
// 6) Flash pass A: per-row softmax stats (max m_i, sum l_i).
//    Wave owns 16 query rows; Q strip staged in LDS (A frags via ds_load),
//    K streamed in 32-token steps (32 WMMA / step, pipelined B fragments).
//    Cross-lane row reductions via 16-wide shfl_xor.  qT pre-scaled.
// ---------------------------------------------------------------------------
__global__ void __launch_bounds__(256)
__attribute__((amdgpu_waves_per_eu(1, 4)))
k_attn_stats(const bf16* qT, const bf16* kT, float* mstat, float* lstat) {
  extern __shared__ __align__(16) char smem_dyn[];
  const int b = blockIdx.y;
  const int wave = threadIdx.x >> 5, lane = threadIdx.x & 31;
  const int hi = lane >> 4;
  const int i0 = blockIdx.x * 128 + wave * 16;
  const bf16* krow = kT + (size_t)b * Nn * Cc;

  bf16* qsh = (bf16*)(smem_dyn + (size_t)wave * STRIP_BYTES);
  stage_strip16(qsh, qT + ((size_t)b * Nn + i0) * Cc);

  float m[8], l[8];
#pragma unroll
  for (int r = 0; r < 8; ++r) { m[r] = -__builtin_inff(); l[r] = 0.f; }

  for (int j0 = 0; j0 < Nn; j0 += 32) {
    if (j0 + 32 < Nn) __builtin_prefetch(krow + (size_t)(j0 + 32) * Cc, 0, 1);
    const bf16* kr0 = krow + (size_t)j0 * Cc;
    const bf16* kr1 = krow + (size_t)(j0 + 16) * Cc;
    v8f s0 = zero8(), s1 = zero8();
    v16bf b0 = load_b_frag(kr0, Cc);
    v16bf b1 = load_b_frag(kr1, Cc);
#pragma unroll
    for (int cs = 0; cs < 16; ++cs) {
      v16bf b0n, b1n;
      if (cs + 1 < 16) {
        b0n = load_b_frag(kr0 + (cs + 1) * 32, Cc);
        b1n = load_b_frag(kr1 + (cs + 1) * 32, Cc);
      }
      v16bf a = load_a_frag(qsh + cs * 32, Cc);          // ds_load, short latency
      s0 = wmma_bf16(a, b0, s0);
      s1 = wmma_bf16(a, b1, s1);
      if (cs + 1 < 16) { b0 = b0n; b1 = b1n; }
    }
#pragma unroll
    for (int r = 0; r < 8; ++r) {
      float a0 = s0[r], a1 = s1[r];
      float mj = fmaxf(a0, a1);
      mj = fmaxf(mj, __shfl_xor(mj, 1, 16));
      mj = fmaxf(mj, __shfl_xor(mj, 2, 16));
      mj = fmaxf(mj, __shfl_xor(mj, 4, 16));
      mj = fmaxf(mj, __shfl_xor(mj, 8, 16));
      float mn = fmaxf(m[r], mj);
      float e  = __expf(a0 - mn) + __expf(a1 - mn);
      e += __shfl_xor(e, 1, 16);
      e += __shfl_xor(e, 2, 16);
      e += __shfl_xor(e, 4, 16);
      e += __shfl_xor(e, 8, 16);
      l[r] = l[r] * __expf(m[r] - mn) + e;
      m[r] = mn;
    }
  }
  if ((lane & 15) == 0) {
    const int rowbase = i0 + 8 * hi;
#pragma unroll
    for (int r = 0; r < 8; ++r) {
      mstat[b * Nn + rowbase + r] = m[r];
      lstat[b * Nn + rowbase + r] = l[r];
    }
  }
}

// ---------------------------------------------------------------------------
// 7) Flash pass B: O^T[i][c] = sum_j (exp(S-m)/l)[i][j] * v[c][j].
//    Wave: 16 query rows x 128-channel slice (64 accum VGPRs).
//    Q strip staged in LDS; S recomputed per c-slice; exp(P) goes through an
//    LDS transpose (D layout -> A layout, bf16); V B-fragments pipelined.
// ---------------------------------------------------------------------------
__global__ void __launch_bounds__(256)
__attribute__((amdgpu_waves_per_eu(1, 4)))
k_attn_out(const bf16* qT, const bf16* kT, const bf16* vbuf, const float* mstat,
           const float* lstat, bf16* oT) {
  extern __shared__ __align__(16) char smem_dyn[];
  __shared__ __align__(16) bf16 pbuf_all[8][16 * 32];  // per-wave P tile (1KB)
  const int b = blockIdx.z;
  const int cslice = blockIdx.y * 128;
  const int wave = threadIdx.x >> 5, lane = threadIdx.x & 31;
  const int hi = lane >> 4, cl = lane & 15;
  const int i0 = blockIdx.x * 128 + wave * 16;

  bf16* pbuf = pbuf_all[wave];
  bf16* qsh  = (bf16*)(smem_dyn + (size_t)wave * STRIP_BYTES);
  stage_strip16(qsh, qT + ((size_t)b * Nn + i0) * Cc);

  const bf16* krow = kT + (size_t)b * Nn * Cc;
  const bf16* vb   = vbuf + (size_t)b * Cc * Nn;

  float m[8], linv[8];
  {
    const int rowbase = i0 + 8 * hi;
#pragma unroll
    for (int r = 0; r < 8; ++r) {
      m[r]    = mstat[b * Nn + rowbase + r];
      linv[r] = 1.f / lstat[b * Nn + rowbase + r];
    }
  }
  v8f acc[8];
#pragma unroll
  for (int ct = 0; ct < 8; ++ct) acc[ct] = zero8();

  for (int j0 = 0; j0 < Nn; j0 += 32) {
    if (j0 + 32 < Nn) __builtin_prefetch(krow + (size_t)(j0 + 32) * Cc, 0, 1);
    const bf16* kr0 = krow + (size_t)j0 * Cc;
    const bf16* kr1 = krow + (size_t)(j0 + 16) * Cc;
    v8f s0 = zero8(), s1 = zero8();
    {
      v16bf b0 = load_b_frag(kr0, Cc);
      v16bf b1 = load_b_frag(kr1, Cc);
#pragma unroll
      for (int cs = 0; cs < 16; ++cs) {
        v16bf b0n, b1n;
        if (cs + 1 < 16) {
          b0n = load_b_frag(kr0 + (cs + 1) * 32, Cc);
          b1n = load_b_frag(kr1 + (cs + 1) * 32, Cc);
        }
        v16bf a = load_a_frag(qsh + cs * 32, Cc);        // ds_load
        s0 = wmma_bf16(a, b0, s0);
        s1 = wmma_bf16(a, b1, s1);
        if (cs + 1 < 16) { b0 = b0n; b1 = b1n; }
      }
    }
    // P = exp(S - m)/l, transposed through LDS into A-fragment layout
#pragma unroll
    for (int r = 0; r < 8; ++r) {
      const int row = r + 8 * hi;
      float p0 = __expf(s0[r] - m[r]) * linv[r];
      float p1 = __expf(s1[r] - m[r]) * linv[r];
      pbuf[row * 32 + cl]      = (bf16)p0;
      pbuf[row * 32 + 16 + cl] = (bf16)p1;
    }
    asm volatile("s_wait_dscnt 0" ::: "memory");        // DS RAW fence (same wave)
    v16bf pa = load_a_frag(pbuf, 32);                   // A[M=i][K=32 j]
    v16bf bb = load_b_frag(vb + (size_t)cslice * Nn + j0, Nn);
#pragma unroll
    for (int ct = 0; ct < 8; ++ct) {
      v16bf bbn;
      if (ct + 1 < 8)
        bbn = load_b_frag(vb + (size_t)(cslice + (ct + 1) * 16) * Nn + j0, Nn);
      acc[ct] = wmma_bf16(pa, bb, acc[ct]);
      if (ct + 1 < 8) bb = bbn;
    }
  }
  // store oT[i][c] bf16 (scattered b16; one-time, negligible traffic)
#pragma unroll
  for (int ct = 0; ct < 8; ++ct) {
    const int c = cslice + ct * 16 + cl;
#pragma unroll
    for (int r = 0; r < 8; ++r)
      oT[((size_t)b * Nn + i0 + r + 8 * hi) * Cc + c] = (bf16)acc[ct][r];
  }
}

// ---------------------------------------------------------------------------
// 8) Output projection + bias + residual:
//    D[M=n][N=co] = oT[n][c] * Wp[co][c]^T; out[b][co][n] = x + D + bp[co].
//    Per-wave A strip staged in LDS; two co-tiles per A fragment; packed
//    float4 loads/stores with residual add.
// ---------------------------------------------------------------------------
__global__ void __launch_bounds__(256)
__attribute__((amdgpu_waves_per_eu(1, 4)))
k_proj_out(const bf16* oT, const bf16* wpb, const float* bp, const float* x,
           float* out) {
  extern __shared__ __align__(16) char smem_dyn[];
  const int b = blockIdx.y;
  const int wave = threadIdx.x >> 5, lane = threadIdx.x & 31;
  const int hi = lane >> 4, cl = lane & 15;
  const int n0 = blockIdx.x * 128 + wave * 16;

  bf16* ash = (bf16*)(smem_dyn + (size_t)wave * STRIP_BYTES);
  stage_strip16(ash, oT + ((size_t)b * Nn + n0) * Cc);

  for (int o0 = 0; o0 < Cc; o0 += 32) {
    v8f acc0 = zero8(), acc1 = zero8();
    v16bf a  = load_a_frag(ash, Cc);
    v16bf b0 = load_b_frag(wpb + (size_t)o0 * Cc, Cc);
    v16bf b1 = load_b_frag(wpb + (size_t)(o0 + 16) * Cc, Cc);
#pragma unroll
    for (int c0 = 0; c0 < Cc; c0 += 32) {
      v16bf an, b0n, b1n;
      if (c0 + 32 < Cc) {
        an  = load_a_frag(ash + c0 + 32, Cc);
        b0n = load_b_frag(wpb + (size_t)o0 * Cc + c0 + 32, Cc);
        b1n = load_b_frag(wpb + (size_t)(o0 + 16) * Cc + c0 + 32, Cc);
      }
      acc0 = wmma_bf16(a, b0, acc0);
      acc1 = wmma_bf16(a, b1, acc1);
      if (c0 + 32 < Cc) { a = an; b0 = b0n; b1 = b1n; }
    }
#pragma unroll
    for (int t = 0; t < 2; ++t) {
      const v8f acc = t ? acc1 : acc0;
      const int oo  = o0 + 16 * t;
      const float bias = bp[oo + cl];
      const size_t base = ((size_t)b * Cc + oo + cl) * Nn + n0 + 8 * hi;
      v4f x0 = *(const v4f*)(x + base);
      v4f x1 = *(const v4f*)(x + base + 4);
      v4f r0, r1;
#pragma unroll
      for (int i = 0; i < 4; ++i) {
        r0[i] = x0[i] + acc[i]     + bias;
        r1[i] = x1[i] + acc[i + 4] + bias;
      }
      *(v4f*)(out + base)     = r0;
      *(v4f*)(out + base + 4) = r1;
    }
  }
}

// ---------------------------------------------------------------------------
// Host launcher.  Workspace layout (bytes), total ~42.1 MB:
//   [mu 256][rstd 256][mstat 32K][lstat 32K][4x bf16 weights 2MB]
//   [hnT 8MB][qT 8MB][kT 8MB][v 8MB][oT 8MB]
// Dynamic LDS: 128KB/block (8 waves x 16KB A strips); gfx1250 allows 320KB.
// ---------------------------------------------------------------------------
extern "C" void kernel_launch(void* const* d_in, const int* in_sizes, int n_in,
                              void* d_out, int out_size, void* d_ws, size_t ws_size,
                              hipStream_t stream) {
  (void)in_sizes; (void)n_in; (void)out_size; (void)ws_size;
  const float* x    = (const float*)d_in[0];
  const float* gn_w = (const float*)d_in[1];
  const float* gn_b = (const float*)d_in[2];
  const float* wq   = (const float*)d_in[3];
  const float* bq   = (const float*)d_in[4];
  const float* wk   = (const float*)d_in[5];
  const float* bk   = (const float*)d_in[6];
  const float* wv   = (const float*)d_in[7];
  const float* bv   = (const float*)d_in[8];
  const float* wp   = (const float*)d_in[9];
  const float* bp   = (const float*)d_in[10];
  float* out = (float*)d_out;

  char* w = (char*)d_ws;
  float* mu    = (float*)w; w += 256;
  float* rstd  = (float*)w; w += 256;
  float* mstat = (float*)w; w += (size_t)Bn * Nn * sizeof(float);
  float* lstat = (float*)w; w += (size_t)Bn * Nn * sizeof(float);
  bf16* wqb = (bf16*)w; w += (size_t)Cc * Cc * sizeof(bf16);
  bf16* wkb = (bf16*)w; w += (size_t)Cc * Cc * sizeof(bf16);
  bf16* wvb = (bf16*)w; w += (size_t)Cc * Cc * sizeof(bf16);
  bf16* wpb = (bf16*)w; w += (size_t)Cc * Cc * sizeof(bf16);
  bf16* hnT = (bf16*)w; w += (size_t)Bn * Nn * Cc * sizeof(bf16);
  bf16* qT  = (bf16*)w; w += (size_t)Bn * Nn * Cc * sizeof(bf16);
  bf16* kT  = (bf16*)w; w += (size_t)Bn * Nn * Cc * sizeof(bf16);
  bf16* vbf = (bf16*)w; w += (size_t)Bn * Cc * Nn * sizeof(bf16);
  bf16* oT  = (bf16*)w; w += (size_t)Bn * Nn * Cc * sizeof(bf16);

  const size_t dynLds = 8 * (size_t)STRIP_BYTES;   // 128KB

  k_w2bf<<<dim3((Cc * Cc + 255) / 256), dim3(256), 0, stream>>>(
      wq, wk, wv, wp, wqb, wkb, wvb, wpb);
  k_gn_stats<<<dim3(Bn * NGRP), dim3(256), 0, stream>>>(x, mu, rstd);
  k_gn_apply<<<dim3((unsigned)((size_t)Bn * Cc * Nn / 256)), dim3(256), 0, stream>>>(
      x, gn_w, gn_b, mu, rstd, hnT);
  k_proj_qk<<<dim3(Nn / 16, Bn), dim3(256), 0, stream>>>(
      hnT, wqb, wkb, bq, bk, qT, kT);
  k_proj_v<<<dim3(Nn / 128, Bn), dim3(256), dynLds, stream>>>(hnT, wvb, bv, vbf);
  k_attn_stats<<<dim3(Nn / 128, Bn), dim3(256), dynLds, stream>>>(
      qT, kT, mstat, lstat);
  k_attn_out<<<dim3(Nn / 128, Cc / 128, Bn), dim3(256), dynLds, stream>>>(
      qT, kT, vbf, mstat, lstat, oT);
  k_proj_out<<<dim3(Nn / 128, Bn), dim3(256), dynLds, stream>>>(
      oT, wpb, bp, x, out);
}
